// SimplifiedHKLFLayer_49873160241880
// MI455X (gfx1250) — matrix operand order repaired and checked by reading, not compiled, not run on hardware
//
#include <hip/hip_runtime.h>
#include <hip/hip_bf16.h>
#include <math.h>

// Problem constants (B=1)
#define Lc   8
#define Cc   16
#define MIDc 8
#define Hc   240
#define Wc   480
#define Rr   8
#define HWc  (Hc * Wc)          // 115200

// Workspace layout (float units). Total = 38,711,040 floats (~155 MB).
#define OFF_HA     ((size_t)0)
#define OFF_HB     ((size_t)14745600)                  // C*H*W*R
#define OFF_MID    ((size_t)29491200)                  // mid as f16: L*MID*HW halves
#define OFF_PHI    (OFF_MID + (size_t)3686400)         // L*HW floats
#define OFF_PSI    (OFF_PHI + (size_t)921600)
#define OFF_FLOW   (OFF_PSI + (size_t)921600)          // L*2*HW floats
#define OFF_XF     (OFF_FLOW + (size_t)1843200)        // C*HW floats
#define OFF_ROWSUM (OFF_XF + (size_t)1843200)          // C*H floats

typedef __attribute__((ext_vector_type(16))) _Float16 v16h;
typedef __attribute__((ext_vector_type(8)))  _Float16 v8h;
typedef __attribute__((ext_vector_type(8)))  float    v8f;

__device__ __forceinline__ float silu_f(float v)    { return v / (1.0f + __expf(-v)); }
__device__ __forceinline__ float sigmoid_f(float v) { return 1.0f / (1.0f + __expf(-v)); }

// ---------------------------------------------------------------------------
// Kernel 1: conv1 (C=16 -> MID=8, 3x3, SAME) + bias + SiLU as implicit GEMM
// via v_wmma_f32_16x16x32_f16. Contraction index K = r*16 + c (r = ky*3+kx),
// padded 144 -> 160 (5 chunks of 32). One block = one row x 128 columns,
// 8 waves x 16 pixels. Input tile staged CHANNEL-LAST in LDS so each B
// fragment is two contiguous ds_load_b128; A fragments likewise.
// ---------------------------------------------------------------------------
__global__ __launch_bounds__(256) void conv1_wmma_kernel(
    const float* __restrict__ x,    // (L,C,H,W)
    const float* __restrict__ w1,   // (MID,C,3,3)
    const float* __restrict__ b1,   // (MID)
    _Float16*    __restrict__ mid)  // (L,MID,H,W) f16
{
    __shared__ _Float16 sA[16 * 160];        // A[m][K], zero-padded, K = r*16+c
    __shared__ _Float16 sX[3 * 132 * 16];    // input tile [dy][j][c] (channel-last)

    const int tid  = threadIdx.x;
    const int lane = tid & 31;
    const int wave = tid >> 5;

    const int bi  = blockIdx.x;
    const int f   = bi / (Hc * 4);
    const int rem = bi % (Hc * 4);
    const int y   = rem >> 2;
    const int x0  = (rem & 3) * 128;

    // Stage A: sA[m*160 + K], K = r*16 + c
    for (int i = tid; i < 16 * 160; i += 256) {
        const int m = i / 160, k = i % 160;
        const int r = k >> 4, c = k & 15;
        float v = 0.0f;
        if (m < MIDc && r < 9) v = w1[(m * Cc + c) * 9 + r];
        sA[i] = (_Float16)v;
    }

    // Stage input tile: 16 ch x 3 rows x 130 cols, zero pad (SAME conv),
    // coalesced global reads per (c,dy) row; channel-last LDS writes.
    const float* xframe = x + (size_t)f * Cc * HWc;
    for (int row = wave; row < 48; row += 8) {
        const int c  = row / 3;
        const int dy = row - c * 3;
        const int yy = y + dy - 1;
        const bool yok = (yy >= 0) && (yy < Hc);
        const float* src = xframe + (size_t)(c * Hc + yy) * Wc;
        for (int j = lane; j < 130; j += 32) {
            const int xx = x0 - 1 + j;
            float v = 0.0f;
            if (yok && xx >= 0 && xx < Wc) v = src[xx];
            sX[(dy * 132 + j) * 16 + c] = (_Float16)v;
        }
    }
    __syncthreads();

    const int n   = lane & 15;
    const int hi  = lane >> 4;
    const int pxl = wave * 16 + n;      // local pixel 0..127
    const int px  = x0 + pxl;

    v8f acc = {};
    #pragma unroll
    for (int kc = 0; kc < 5; ++kc) {
        // A fragment: 16 contiguous halves at K = kbaseA..+15 (matches B).
        const int kbaseA = kc * 32 + 16 * hi;
        const _Float16* arow = sA + n * 160 + kbaseA;
        const v8h a_lo = *(const v8h*)(arow);
        const v8h a_hi = *(const v8h*)(arow + 8);
        const v16h a = __builtin_shufflevector(a_lo, a_hi,
            0, 1, 2, 3, 4, 5, 6, 7, 8, 9, 10, 11, 12, 13, 14, 15);

        // B fragment: K = kbaseB + e, kbaseB multiple of 16 -> kernel pos r
        // constant per lane, c == e -> 16 contiguous halves (channel-last).
        const int rB = kc * 2 + hi;     // kernel position 0..9 (9 == padding)
        v16h b = {};
        if (rB < 9) {
            const int dy = rB / 3;
            const int dx = rB - dy * 3;
            const _Float16* brow = sX + (size_t)(dy * 132 + pxl + dx) * 16;
            const v8h b_lo = *(const v8h*)(brow);
            const v8h b_hi = *(const v8h*)(brow + 8);
            b = __builtin_shufflevector(b_lo, b_hi,
                0, 1, 2, 3, 4, 5, 6, 7, 8, 9, 10, 11, 12, 13, 14, 15);
        }
        acc = __builtin_amdgcn_wmma_f32_16x16x32_f16(
            /*neg_a=*/false, a, /*neg_b=*/false, b,
            /*c_mod=*/(short)0, acc, /*reuse_a=*/false, /*reuse_b=*/false);
    }

    // D layout: lanes 0-15 hold M = r (0..7) == the 8 valid output channels.
    if (lane < 16 && px < Wc) {
        #pragma unroll
        for (int r = 0; r < 8; ++r) {
            float v = silu_f(acc[r] + b1[r]);
            mid[((size_t)f * MIDc + r) * HWc + y * Wc + px] = (_Float16)v;
        }
    }
}

// ---------------------------------------------------------------------------
// Kernel 2: conv2 (MID=8 -> 2, 3x3, SAME) + bias -> phi, psi. LDS-tiled,
// branch-free inner loop. One block = one row x 256 columns.
// ---------------------------------------------------------------------------
__global__ __launch_bounds__(256) void conv2_kernel(
    const _Float16* __restrict__ mid,   // (L,MID,H,W) f16
    const float* __restrict__ w2,       // (2,MID,3,3)
    const float* __restrict__ b2,       // (2)
    float* __restrict__ phi,            // (L,H,W)
    float* __restrict__ psi)            // (L,H,W)
{
    __shared__ _Float16 sM[8 * 3 * 258];   // mid tile [m][dy][j], j: x0-1..x0+256
    __shared__ float    sW[146];           // 144 weights + 2 biases

    const int tid  = threadIdx.x;
    const int lane = tid & 31;
    const int wave = tid >> 5;
    const int bi   = blockIdx.x;
    const int f    = bi / (Hc * 2);
    const int rem  = bi % (Hc * 2);
    const int y    = rem >> 1;
    const int x0   = (rem & 1) * 256;

    for (int i = tid; i < 146; i += 256) sW[i] = (i < 144) ? w2[i] : b2[i - 144];

    const _Float16* mf = mid + (size_t)f * MIDc * HWc;
    for (int row = wave; row < 24; row += 8) {
        const int m  = row / 3;
        const int dy = row - m * 3;
        const int yy = y + dy - 1;
        const bool yok = (yy >= 0) && (yy < Hc);
        const _Float16* src = mf + (size_t)(m * Hc + yy) * Wc;
        _Float16* dstrow = sM + (m * 3 + dy) * 258;
        for (int j = lane; j < 258; j += 32) {
            const int xx = x0 - 1 + j;
            _Float16 v = (_Float16)0.0f;
            if (yok && xx >= 0 && xx < Wc) v = src[xx];
            dstrow[j] = v;
        }
    }
    __syncthreads();

    const int px = x0 + tid;
    if (px >= Wc) return;

    float a0 = sW[144], a1 = sW[145];
    #pragma unroll
    for (int m = 0; m < MIDc; ++m) {
        #pragma unroll
        for (int dy = 0; dy < 3; ++dy) {
            const int base = (m * 3 + dy) * 258 + tid;
            #pragma unroll
            for (int dx = 0; dx < 3; ++dx) {
                const float v  = (float)sM[base + dx];
                const int   wi = (m * 3 + dy) * 3 + dx;
                a0 += sW[wi] * v;
                a1 += sW[72 + wi] * v;
            }
        }
    }
    phi[(size_t)f * HWc + y * Wc + px] = a0;
    psi[(size_t)f * HWc + y * Wc + px] = a1;
}

// ---------------------------------------------------------------------------
// Kernel 3: Sobel gradients (wrap pad x, edge pad y) + metric + tanh -> flow
// ---------------------------------------------------------------------------
__global__ __launch_bounds__(256) void sobel_flow_kernel(
    const float* __restrict__ phi,
    const float* __restrict__ psi,
    float* __restrict__ flow)           // (L,2,H,W)
{
    const int idx = blockIdx.x * 256 + threadIdx.x;
    if (idx >= Lc * HWc) return;
    const int f = idx / HWc, p = idx % HWc;
    const int py = p / Wc, px = p % Wc;
    const float* ph = phi + (size_t)f * HWc;
    const float* ps = psi + (size_t)f * HWc;

    float phv[3][3], psv[3][3];
    #pragma unroll
    for (int dy = -1; dy <= 1; ++dy) {
        int yy = py + dy;
        yy = yy < 0 ? 0 : (yy >= Hc ? Hc - 1 : yy);   // edge pad y
        #pragma unroll
        for (int dx = -1; dx <= 1; ++dx) {
            int xx = px + dx;                          // wrap pad x (no modulo)
            xx += (xx < 0) ? Wc : 0;
            xx -= (xx >= Wc) ? Wc : 0;
            phv[dy + 1][dx + 1] = ph[yy * Wc + xx];
            psv[dy + 1][dx + 1] = ps[yy * Wc + xx];
        }
    }
    const float gx_phi = -phv[0][0] + phv[0][2] - 2.f * phv[1][0] + 2.f * phv[1][2] - phv[2][0] + phv[2][2];
    const float gy_phi = -phv[0][0] - 2.f * phv[0][1] - phv[0][2] + phv[2][0] + 2.f * phv[2][1] + phv[2][2];
    const float gx_psi = -psv[0][0] + psv[0][2] - 2.f * psv[1][0] + 2.f * psv[1][2] - psv[2][0] + psv[2][2];
    const float gy_psi = -psv[0][0] - 2.f * psv[0][1] - psv[0][2] + psv[2][0] + 2.f * psv[2][1] + psv[2][2];

    const float lat    = -1.5707963267948966f + 3.14159265358979323f * (float)py / (float)(Hc - 1);
    const float metric = 1.0f / (__cosf(lat) + 1e-6f);
    const float u = gx_phi * metric - gy_psi;
    const float v = gy_phi + gx_psi * metric;
    flow[((size_t)f * 2 + 0) * HWc + p] = tanhf(u);
    flow[((size_t)f * 2 + 1) * HWc + p] = tanhf(v);
}

// ---------------------------------------------------------------------------
// Kernel 4: one scan step: bilinear warp (border clamp) of h, + x_t,
// xf = mean over R, and deterministic per-row sum. One block per (c,y) row;
// R=8 fiber -> each corner = two coalesced float4 loads.
// ---------------------------------------------------------------------------
__global__ __launch_bounds__(256) void warp_step_kernel(
    const float* __restrict__ h_old,    // (C,H,W,R)
    float*       __restrict__ h_new,    // (C,H,W,R)
    const float* __restrict__ flow,     // (L,2,H,W)
    const float* __restrict__ x,        // (L,C,H,W)
    const float* __restrict__ listT,    // (L)
    int t,
    float* __restrict__ xf,             // (C,H,W)
    float* __restrict__ rowsum)         // (C,H)
{
    const int c   = blockIdx.x / Hc;
    const int y   = blockIdx.x % Hc;
    const int tid = threadIdx.x;
    const float dt = listT[t];
    const float gy = -1.0f + 2.0f * (float)y / (float)(Hc - 1);

    const float* fu   = flow + ((size_t)t * 2 + 0) * HWc + y * Wc;
    const float* fv   = flow + ((size_t)t * 2 + 1) * HWc + y * Wc;
    const float* xrow = x + (((size_t)t * Cc + c) * Hc + y) * (size_t)Wc;

    float localsum = 0.0f;
    for (int xi = tid; xi < Wc; xi += 256) {
        const float gx  = -1.0f + 2.0f * (float)xi / (float)(Wc - 1);
        const float gxs = gx - fu[xi] * dt;
        const float gys = gy - fv[xi] * dt;
        float xp = (gxs + 1.0f) * (Wc * 0.5f) - 0.5f;
        float yp = (gys + 1.0f) * (Hc * 0.5f) - 0.5f;
        xp = fminf(fmaxf(xp, 0.0f), (float)(Wc - 1));
        yp = fminf(fmaxf(yp, 0.0f), (float)(Hc - 1));
        const float x0f = floorf(xp), y0f = floorf(yp);
        const float wx = xp - x0f, wy = yp - y0f;
        const int x0 = (int)x0f, y0 = (int)y0f;
        const int x1 = min(x0 + 1, Wc - 1);
        const int y1 = min(y0 + 1, Hc - 1);

        const float* b00 = h_old + (((size_t)c * Hc + y0) * Wc + x0) * Rr;
        const float* b01 = h_old + (((size_t)c * Hc + y0) * Wc + x1) * Rr;
        const float* b10 = h_old + (((size_t)c * Hc + y1) * Wc + x0) * Rr;
        const float* b11 = h_old + (((size_t)c * Hc + y1) * Wc + x1) * Rr;
        float* dst = h_new + (((size_t)c * Hc + y) * Wc + xi) * Rr;

        const float xadd = xrow[xi];
        const float cwx = 1.0f - wx, cwy = 1.0f - wy;
        float s = 0.0f;
        #pragma unroll
        for (int h4 = 0; h4 < 2; ++h4) {
            const float4 v00 = *(const float4*)(b00 + h4 * 4);
            const float4 v01 = *(const float4*)(b01 + h4 * 4);
            const float4 v10 = *(const float4*)(b10 + h4 * 4);
            const float4 v11 = *(const float4*)(b11 + h4 * 4);
            float4 o;
            o.x = (v00.x * cwx + v01.x * wx) * cwy + (v10.x * cwx + v11.x * wx) * wy + xadd;
            o.y = (v00.y * cwx + v01.y * wx) * cwy + (v10.y * cwx + v11.y * wx) * wy + xadd;
            o.z = (v00.z * cwx + v01.z * wx) * cwy + (v10.z * cwx + v11.z * wx) * wy + xadd;
            o.w = (v00.w * cwx + v01.w * wx) * cwy + (v10.w * cwx + v11.w * wx) * wy + xadd;
            *(float4*)(dst + h4 * 4) = o;
            s += o.x + o.y + o.z + o.w;
        }
        const float m = s * 0.125f;
        xf[(size_t)c * HWc + y * Wc + xi] = m;
        localsum += m;
    }

    __shared__ float red[256];
    red[tid] = localsum;
    __syncthreads();
    #pragma unroll
    for (int s2 = 128; s2 > 0; s2 >>= 1) {
        if (tid < s2) red[tid] += red[tid + s2];
        __syncthreads();
    }
    if (tid == 0) rowsum[c * Hc + y] = red[0];
}

// ---------------------------------------------------------------------------
// Kernel 5: SE gate (16 -> 4 SiLU -> 16 sigmoid, per (c,y) row) and output.
// ---------------------------------------------------------------------------
__global__ __launch_bounds__(256) void gate_out_kernel(
    const float* __restrict__ xf,       // (C,H,W)
    const float* __restrict__ rowsum,   // (C,H)
    const float* __restrict__ sw1,      // (4,16)
    const float* __restrict__ sb1,      // (4)
    const float* __restrict__ sw2,      // (16,4)
    const float* __restrict__ sb2,      // (16)
    float* __restrict__ out)            // (C,H,W) slice for step t
{
    const int c   = blockIdx.x / Hc;
    const int y   = blockIdx.x % Hc;
    const int tid = threadIdx.x;

    float rm[Cc];
    #pragma unroll
    for (int cc = 0; cc < Cc; ++cc) rm[cc] = rowsum[cc * Hc + y] * (1.0f / (float)Wc);

    float z[4];
    #pragma unroll
    for (int m = 0; m < 4; ++m) {
        float a = sb1[m];
        #pragma unroll
        for (int cc = 0; cc < Cc; ++cc) a += sw1[m * Cc + cc] * rm[cc];
        z[m] = silu_f(a);
    }
    float g = sb2[c];
    #pragma unroll
    for (int m = 0; m < 4; ++m) g += sw2[c * 4 + m] * z[m];
    g = sigmoid_f(g);

    const float* xrow = xf + (size_t)c * HWc + y * Wc;
    float* orow = out + (size_t)c * HWc + y * Wc;
    for (int xi = tid; xi < Wc; xi += 256) orow[xi] = xrow[xi] * g;
}

// ---------------------------------------------------------------------------
extern "C" void kernel_launch(void* const* d_in, const int* in_sizes, int n_in,
                              void* d_out, int out_size, void* d_ws, size_t ws_size,
                              hipStream_t stream) {
    (void)in_sizes; (void)n_in; (void)out_size; (void)ws_size;
    const float* x     = (const float*)d_in[0];
    const float* h0    = (const float*)d_in[1];
    const float* listT = (const float*)d_in[2];
    const float* w1    = (const float*)d_in[3];
    const float* b1    = (const float*)d_in[4];
    const float* w2    = (const float*)d_in[5];
    const float* b2    = (const float*)d_in[6];
    const float* sw1   = (const float*)d_in[7];
    const float* sb1   = (const float*)d_in[8];
    const float* sw2   = (const float*)d_in[9];
    const float* sb2   = (const float*)d_in[10];
    float* out = (float*)d_out;
    float* ws  = (float*)d_ws;

    float*    hA     = ws + OFF_HA;
    float*    hB     = ws + OFF_HB;
    _Float16* mid    = (_Float16*)(ws + OFF_MID);
    float*    phi    = ws + OFF_PHI;
    float*    psi    = ws + OFF_PSI;
    float*    flow   = ws + OFF_FLOW;
    float*    xf     = ws + OFF_XF;
    float*    rowsum = ws + OFF_ROWSUM;

    // Flow field (all L frames in parallel)
    conv1_wmma_kernel<<<Lc * Hc * 4, 256, 0, stream>>>(x, w1, b1, mid);
    conv2_kernel<<<Lc * Hc * 2, 256, 0, stream>>>(mid, w2, b2, phi, psi);
    sobel_flow_kernel<<<(Lc * HWc + 255) / 256, 256, 0, stream>>>(phi, psi, flow);

    // Sequential scan over L steps, ping-pong h in workspace
    const float* hold = h0;
    float*       hnew = hA;
    for (int t = 0; t < Lc; ++t) {
        warp_step_kernel<<<Cc * Hc, 256, 0, stream>>>(hold, hnew, flow, x, listT, t, xf, rowsum);
        gate_out_kernel<<<Cc * Hc, 256, 0, stream>>>(xf, rowsum, sw1, sb1, sw2, sb2,
                                                     out + (size_t)t * Cc * HWc);
        hold = hnew;
        hnew = (hnew == hA) ? hB : hA;
    }
}
